// BiLSTM_CRF_21638045237625
// MI455X (gfx1250) — compile-verified, matrix operand order checked by reading, not run on hardware
//
#include <hip/hip_runtime.h>
#include <hip/hip_bf16.h>
#include <math.h>

typedef __attribute__((ext_vector_type(16))) __bf16 v16bf;
typedef __attribute__((ext_vector_type(8)))  float  v8f;

#define B_  8
#define S_  512
#define E_  512
#define HD_ 512
#define G4_ 2048          // 4*HD
#define H_  1024
#define T_  52
#define TPAD_ 64
#define MROWS_ 4096       // B*S

static __device__ __forceinline__ unsigned short f32_to_bf16_bits(float f) {
    union { float f; unsigned int u; } v; v.f = f;
    unsigned int u = v.u;
    // round-to-nearest-even truncation to bf16
    unsigned int r = (u + 0x7FFFu + ((u >> 16) & 1u)) >> 16;
    return (unsigned short)r;
}

// ---------------- conversion / gather kernels ----------------

__global__ void k_f32_to_bf16(const float* __restrict__ in,
                              unsigned short* __restrict__ out, int n) {
    int i = blockIdx.x * blockDim.x + threadIdx.x;
    if (i < n) out[i] = f32_to_bf16_bits(in[i]);
}

// fc_W [52,1024] f32 -> [64,1024] bf16, rows 52..63 zero
__global__ void k_convert_fcw(const float* __restrict__ fcW,
                              unsigned short* __restrict__ out) {
    int i = blockIdx.x * blockDim.x + threadIdx.x;
    if (i < TPAD_ * H_) {
        int r = i >> 10;
        out[i] = (r < T_) ? f32_to_bf16_bits(fcW[i]) : (unsigned short)0;
    }
}

// emb_bf[row, c] = bf16(table[X[row], c]), row = b*512+s
__global__ void k_gather_emb(const int* __restrict__ X,
                             const float* __restrict__ table,
                             unsigned short* __restrict__ out) {
    int i = blockIdx.x * blockDim.x + threadIdx.x;
    if (i < MROWS_ * E_) {
        int row = i >> 9;           // /512
        int c   = i & (E_ - 1);
        int tok = X[row];
        out[i] = f32_to_bf16_bits(table[(size_t)tok * E_ + c]);
    }
}

// hBf = bf16(h0dir), cSt = c0dir
__global__ void k_init_state(const float* __restrict__ h0dir,
                             const float* __restrict__ c0dir,
                             unsigned short* __restrict__ hBf,
                             float* __restrict__ cSt) {
    int i = blockIdx.x * blockDim.x + threadIdx.x;
    if (i < S_ * HD_) {
        hBf[i] = f32_to_bf16_bits(h0dir[i]);
        cSt[i] = c0dir[i];
    }
}

// ---------------- WMMA GEMM: C[M,N] = A[M,K] @ Bw[N,K]^T (+bias) ----------------
// A, Bw bf16 row-major; one wave (32 threads) computes a 32x64 tile with
// 2x4 v_wmma_f32_16x16x32_bf16 accumulators (8 WMMA per 6 fragment loads,
// ~21 FLOP/byte from L2).
__global__ void k_wmma_gemm(const unsigned short* __restrict__ A,
                            const unsigned short* __restrict__ Bw,
                            float* __restrict__ C,
                            const float* __restrict__ bias, int biasN,
                            int K, int ldc) {
    int lane = threadIdx.x;           // blockDim.x == 32
    int n0 = blockIdx.x * 64;
    int m0 = blockIdx.y * 32;
    int lr = lane & 15;               // row within 16-tile
    int kh = lane >> 4;               // which 16-wide K half this lane holds

    const unsigned short* a0p = A  + (size_t)(m0 + lr) * K + kh * 16;
    const unsigned short* a1p = a0p + (size_t)16 * K;
    const unsigned short* b0p = Bw + (size_t)(n0 + lr) * K + kh * 16;
    const unsigned short* b1p = b0p + (size_t)16 * K;
    const unsigned short* b2p = b0p + (size_t)32 * K;
    const unsigned short* b3p = b0p + (size_t)48 * K;

    v8f acc[2][4] = {};
    for (int k = 0; k < K; k += 32) {
        v16bf a0 = *(const v16bf*)(a0p + k);
        v16bf a1 = *(const v16bf*)(a1p + k);
        v16bf b0 = *(const v16bf*)(b0p + k);
        v16bf b1 = *(const v16bf*)(b1p + k);
        v16bf b2 = *(const v16bf*)(b2p + k);
        v16bf b3 = *(const v16bf*)(b3p + k);
        acc[0][0] = __builtin_amdgcn_wmma_f32_16x16x32_bf16(false, a0, false, b0, (short)0, acc[0][0], false, false);
        acc[0][1] = __builtin_amdgcn_wmma_f32_16x16x32_bf16(false, a0, false, b1, (short)0, acc[0][1], false, false);
        acc[0][2] = __builtin_amdgcn_wmma_f32_16x16x32_bf16(false, a0, false, b2, (short)0, acc[0][2], false, false);
        acc[0][3] = __builtin_amdgcn_wmma_f32_16x16x32_bf16(false, a0, false, b3, (short)0, acc[0][3], false, false);
        acc[1][0] = __builtin_amdgcn_wmma_f32_16x16x32_bf16(false, a1, false, b0, (short)0, acc[1][0], false, false);
        acc[1][1] = __builtin_amdgcn_wmma_f32_16x16x32_bf16(false, a1, false, b1, (short)0, acc[1][1], false, false);
        acc[1][2] = __builtin_amdgcn_wmma_f32_16x16x32_bf16(false, a1, false, b2, (short)0, acc[1][2], false, false);
        acc[1][3] = __builtin_amdgcn_wmma_f32_16x16x32_bf16(false, a1, false, b3, (short)0, acc[1][3], false, false);
    }

    int cm = lane >> 4;               // +8 row offset for lanes 16..31
    int cn = lane & 15;
    #pragma unroll
    for (int mt = 0; mt < 2; ++mt) {
        #pragma unroll
        for (int nt = 0; nt < 4; ++nt) {
            int col = n0 + nt * 16 + cn;
            float bv = (bias != nullptr && col < biasN) ? bias[col] : 0.0f;
            #pragma unroll
            for (int v = 0; v < 8; ++v) {
                int row = m0 + mt * 16 + v + 8 * cm;
                C[(size_t)row * ldc + col] = acc[mt][nt][v] + bv;
            }
        }
    }
}

// ---------------- fused LSTM gate elementwise ----------------
// Gin/Gh: [512, 2048]; torch gate order i,f,g,o in chunks of 512.
__global__ void k_lstm_gate(const float* __restrict__ Gin,
                            const float* __restrict__ Gh,
                            const float* __restrict__ bias,
                            float* __restrict__ cSt,
                            unsigned short* __restrict__ hBf,
                            unsigned short* __restrict__ lstmBf,
                            int rowBase, int colBase) {
    int idx = blockIdx.x * blockDim.x + threadIdx.x;
    if (idx >= S_ * HD_) return;
    int s = idx >> 9;                 // /512
    int j = idx & (HD_ - 1);
    size_t gb = (size_t)s * G4_ + j;
    float gi = Gin[gb]            + Gh[gb]            + bias[j];
    float gf = Gin[gb + 512]      + Gh[gb + 512]      + bias[j + 512];
    float gg = Gin[gb + 1024]     + Gh[gb + 1024]     + bias[j + 1024];
    float go = Gin[gb + 1536]     + Gh[gb + 1536]     + bias[j + 1536];
    float si = 1.0f / (1.0f + __expf(-gi));
    float sf = 1.0f / (1.0f + __expf(-gf));
    float so = 1.0f / (1.0f + __expf(-go));
    float c  = sf * cSt[idx] + si * tanhf(gg);
    float h  = so * tanhf(c);
    cSt[idx] = c;
    unsigned short hb = f32_to_bf16_bits(h);
    hBf[idx] = hb;
    lstmBf[(size_t)(rowBase + s) * H_ + colBase + j] = hb;
}

// ---------------- CRF forward + gold score (single block) ----------------
__global__ void k_crf(const float* __restrict__ feats,   // [4096, 64]
                      const float* __restrict__ trans,   // [52, 52]
                      const int*   __restrict__ Y,       // [8, 512]
                      float* __restrict__ out) {
    __shared__ float sTr[T_ * T_];
    __shared__ float fvA[B_ * T_];
    __shared__ float fvB[B_ * T_];
    int tid = threadIdx.x;            // 512 threads
    for (int i = tid; i < T_ * T_; i += blockDim.x) sTr[i] = trans[i];
    if (tid < B_ * T_) fvA[tid] = ((tid % T_) == (T_ - 2)) ? 0.0f : -1.0e4f;
    __syncthreads();

    float* cur = fvA;
    float* nxt = fvB;
    int b  = tid >> 6;                // /64 -> 0..7
    int nx = tid & 63;
    bool act = (nx < T_);
    for (int s = 0; s < S_; ++s) {
        if (act) {
            float feat = feats[(size_t)((b << 9) + s) * TPAD_ + nx];
            const float* trow = sTr + nx * T_;
            const float* fvb  = cur + b * T_;
            float m = -INFINITY;
            for (int p = 0; p < T_; ++p) m = fmaxf(m, fvb[p] + trow[p]);
            float ssum = 0.0f;
            for (int p = 0; p < T_; ++p) ssum += __expf(fvb[p] + trow[p] - m);
            nxt[b * T_ + nx] = m + __logf(ssum) + feat;
        }
        __syncthreads();
        float* t = cur; cur = nxt; nxt = t;
    }
    __syncthreads();
    if (tid == 0) {
        float fs = 0.0f;
        for (int bb = 0; bb < B_; ++bb) {
            float m = -INFINITY;
            for (int t = 0; t < T_; ++t)
                m = fmaxf(m, cur[bb * T_ + t] + sTr[(T_ - 1) * T_ + t]);
            float ssum = 0.0f;
            for (int t = 0; t < T_; ++t)
                ssum += __expf(cur[bb * T_ + t] + sTr[(T_ - 1) * T_ + t] - m);
            fs += m + __logf(ssum);
        }
        float gs = 0.0f;
        for (int bb = 0; bb < B_; ++bb) {
            int prev = T_ - 2;
            for (int s = 0; s < S_; ++s) {
                int tg = Y[bb * S_ + s];
                gs += sTr[tg * T_ + prev];
                gs += feats[(size_t)((bb << 9) + s) * TPAD_ + tg];
                prev = tg;
            }
            gs += sTr[(T_ - 1) * T_ + prev];
        }
        out[0] = fs - gs;
    }
}

__global__ void k_copy_y(const int* __restrict__ Y, float* __restrict__ out, int n) {
    int i = blockIdx.x * blockDim.x + threadIdx.x;
    if (i < n) out[1 + i] = (float)Y[i];
}

// ---------------- launcher ----------------
extern "C" void kernel_launch(void* const* d_in, const int* in_sizes, int n_in,
                              void* d_out, int out_size, void* d_ws, size_t ws_size,
                              hipStream_t stream) {
    const int*   X     = (const int*)  d_in[0];
    const int*   Y     = (const int*)  d_in[1];
    const float* h0    = (const float*)d_in[2];   // [2,512,512]
    const float* c0    = (const float*)d_in[3];
    const float* emb   = (const float*)d_in[4];   // [50257,512]
    const float* Wih_f = (const float*)d_in[5];   // [2048,512]
    const float* Whh_f = (const float*)d_in[6];   // [2048,512]
    const float* b_f   = (const float*)d_in[7];
    const float* Wih_b = (const float*)d_in[8];
    const float* Whh_b = (const float*)d_in[9];
    const float* b_b   = (const float*)d_in[10];
    const float* fc_W  = (const float*)d_in[11];  // [52,1024]
    const float* fc_b  = (const float*)d_in[12];
    const float* trans = (const float*)d_in[13];  // [52,52]
    float* out = (float*)d_out;

    // workspace layout (bytes, 256-aligned)
    char* w = (char*)d_ws;
    unsigned short* embBf  = (unsigned short*)(w + 0);                 // 4096*512  bf16
    unsigned short* wihBf  = (unsigned short*)(w + 4194304);           // 2048*512  bf16
    unsigned short* whhBf  = (unsigned short*)(w + 6291456);           // 2048*512  bf16
    unsigned short* fcWBf  = (unsigned short*)(w + 8388608);           // 64*1024   bf16
    float*          Gin    = (float*)(w + 8519680);                    // 4096*2048 f32
    float*          Gh     = (float*)(w + 42074112);                   // 512*2048  f32
    float*          cSt    = (float*)(w + 46268416);                   // 512*512   f32
    unsigned short* hBf    = (unsigned short*)(w + 47316992);          // 512*512   bf16
    unsigned short* lstmBf = (unsigned short*)(w + 47841280);          // 4096*1024 bf16
    float*          feats  = (float*)(w + 56229888);                   // 4096*64   f32

    const int TB = 256;

    // fc weights -> bf16 padded [64,1024]
    k_convert_fcw<<<(TPAD_ * H_ + TB - 1) / TB, TB, 0, stream>>>(fc_W, fcWBf);
    // embedding gather + bf16
    k_gather_emb<<<(MROWS_ * E_ + TB - 1) / TB, TB, 0, stream>>>(X, emb, embBf);

    for (int dir = 0; dir < 2; ++dir) {
        const float* Wih = dir ? Wih_b : Wih_f;
        const float* Whh = dir ? Whh_b : Whh_f;
        const float* bb  = dir ? b_b   : b_f;

        k_f32_to_bf16<<<(G4_ * E_ + TB - 1) / TB, TB, 0, stream>>>(Wih, wihBf, G4_ * E_);
        k_f32_to_bf16<<<(G4_ * HD_ + TB - 1) / TB, TB, 0, stream>>>(Whh, whhBf, G4_ * HD_);

        // all-step input projections: Gin[4096,2048] = embBf @ Wih^T
        {
            dim3 grid(G4_ / 64, MROWS_ / 32);
            k_wmma_gemm<<<grid, 32, 0, stream>>>(embBf, wihBf, Gin, nullptr, 0, E_, G4_);
        }
        // initial state
        k_init_state<<<(S_ * HD_ + TB - 1) / TB, TB, 0, stream>>>(
            h0 + (size_t)dir * S_ * HD_, c0 + (size_t)dir * S_ * HD_, hBf, cSt);

        for (int t = 0; t < B_; ++t) {
            int tmem = dir ? (B_ - 1 - t) : t;   // input time == output row (reverse+re-reverse)
            // Gh[512,2048] = hBf @ Whh^T
            dim3 grid(G4_ / 64, S_ / 32);
            k_wmma_gemm<<<grid, 32, 0, stream>>>(hBf, whhBf, Gh, nullptr, 0, HD_, G4_);
            k_lstm_gate<<<(S_ * HD_ + TB - 1) / TB, TB, 0, stream>>>(
                Gin + (size_t)tmem * S_ * G4_, Gh, bb, cSt, hBf, lstmBf,
                tmem * S_, dir * HD_);
        }
    }

    // feats[4096,64] = lstmBf[4096,1024] @ fcWBf[64,1024]^T + fc_b
    {
        dim3 grid(TPAD_ / 64, MROWS_ / 32);
        k_wmma_gemm<<<grid, 32, 0, stream>>>(lstmBf, fcWBf, feats, fc_b, T_, H_, TPAD_);
    }

    // CRF forward + gold score -> out[0]
    k_crf<<<1, 512, 0, stream>>>(feats, trans, Y, out);
    // pass-through Y -> out[1..4096]
    k_copy_y<<<(B_ * S_ + TB - 1) / TB, TB, 0, stream>>>(Y, out, B_ * S_);
}
